// MSDeformAttnHead_63668595196601
// MI455X (gfx1250) — matrix operand order repaired and checked by reading, not compiled, not run on hardware
//
#include <hip/hip_runtime.h>
#include <math.h>

// ---------------------------------------------------------------------------
// CDNA5 (gfx1250) WMMA types
// ---------------------------------------------------------------------------
typedef __attribute__((ext_vector_type(16))) __bf16 v16bf;
typedef __attribute__((ext_vector_type(8)))  __bf16 v8bf;
typedef __attribute__((ext_vector_type(4)))  __bf16 v4bf;
typedef __attribute__((ext_vector_type(8)))  float  v8f;

__device__ __forceinline__ __bf16 f2bf(float f) {
  union { float f; unsigned u; } v; v.f = f;
  unsigned r = (v.u + 0x7FFFu + ((v.u >> 16) & 1u)) >> 16;  // round-nearest-even
  union { unsigned short s; __bf16 b; } o; o.s = (unsigned short)r;
  return o.b;
}

// f32 -> bf16, same layout, 4 elems/thread (b128 in, b64 out)
__global__ __launch_bounds__(256)
void k_cvt_bf16(const float* __restrict__ X, __bf16* __restrict__ Y, int total)
{
  int i = (blockIdx.x * 256 + threadIdx.x) * 4;
  if (i >= total) return;
  const float4 f = *(const float4*)(X + i);
  v4bf o; o[0] = f2bf(f.x); o[1] = f2bf(f.y); o[2] = f2bf(f.z); o[3] = f2bf(f.w);
  *(v4bf*)(Y + i) = o;
}

// f32 (R,C) -> bf16 (C,R) transpose+convert (coalesced writes)
__global__ __launch_bounds__(256)
void k_cvt_t_bf16(const float* __restrict__ X, __bf16* __restrict__ Y, int R, int C)
{
  int i = blockIdx.x * 256 + threadIdx.x;
  if (i >= R * C) return;
  const int c = i / R, r = i - c * R;
  Y[i] = f2bf(X[(long)r * C + c]);
}

// ---------------------------------------------------------------------------
// Fast GEMM: D[m,n] = sum_k A[m,k]*Bt[n,k] (+bias) (+res) (relu)
// A: bf16 (M,K) row-major. Bt: bf16 (N,K) row-major (i.e. W^T).
// One wave = one 16x16 tile; K stepped by 64 (2 WMMAs / iter); fragments are
// pure b128 loads + shufflevector. M,N mult of 16; K mult of 64.
// ---------------------------------------------------------------------------
__global__ __launch_bounds__(256)
void k_wmma_gemm_bf(const __bf16* __restrict__ A, const __bf16* __restrict__ Bt,
                    const float* __restrict__ bias, int bias_per_m,
                    const float* __restrict__ res, long r_ms, long r_ns,
                    float* __restrict__ D, long d_ms, long d_ns,
                    int M, int N, int K, int relu)
{
  const int wave = threadIdx.x >> 5;
  const int lane = threadIdx.x & 31;
  const int n0 = (blockIdx.x * 8 + wave) << 4;
  const int m0 = (int)blockIdx.y << 4;
  if (n0 >= N) return;                     // wave-uniform (EXEC all-ones for WMMA)
  const int half = lane >> 4;
  const int l16  = lane & 15;
  const __bf16* arow = A  + (long)(m0 + l16) * K;   // A row this lane holds
  const __bf16* brow = Bt + (long)(n0 + l16) * K;   // W^T row (= W column)
  v8f acc = {};
  for (int k0 = 0; k0 < K; k0 += 64) {
    if (k0 + 128 < K) {
      __builtin_prefetch(arow + k0 + 128, 0, 1);
      __builtin_prefetch(brow + k0 + 128, 0, 1);
    }
#pragma unroll
    for (int u = 0; u < 2; ++u) {
      const int kk = k0 + u * 32;
      // A 16x32 bf16, k order used here (same as B): lane half 0 -> K 0..15, half 1 -> K 16..31
      v8bf a0 = *(const v8bf*)(arow + kk + (half << 4));
      v8bf a1 = *(const v8bf*)(arow + kk + (half << 4) + 8);
      // B 32x16 bf16, k order used here (same as A): lane half 0 -> K 0..15, half 1 -> K 16..31
      v8bf b0 = *(const v8bf*)(brow + kk + (half << 4));
      v8bf b1 = *(const v8bf*)(brow + kk + (half << 4) + 8);
      v16bf a = __builtin_shufflevector(a0, a1, 0,1,2,3,4,5,6,7,8,9,10,11,12,13,14,15);
      v16bf b = __builtin_shufflevector(b0, b1, 0,1,2,3,4,5,6,7,8,9,10,11,12,13,14,15);
      acc = __builtin_amdgcn_wmma_f32_16x16x32_bf16(false, a, false, b,
                                                    (short)0, acc, false, false);
    }
  }
#pragma unroll
  for (int r = 0; r < 8; ++r) {
    const int cm = m0 + r + (half << 3);
    const int cn = n0 + l16;
    float v = acc[r];
    if (bias) v += bias[bias_per_m ? cm : cn];
    if (res)  v += res[(long)cm * r_ms + (long)cn * r_ns];
    if (relu) v = fmaxf(v, 0.f);
    D[(long)cm * d_ms + (long)cn * d_ns] = v;
  }
}

// ---------------------------------------------------------------------------
// 3x3 conv (pad 1) as implicit GEMM. M = cout (A = bf16 weights (256,2304),
// vector loads), N = pixels (B gathered bf16 from pixel-major image with halo
// bounds), K = 2304. Output NCHW f32, one batch image per launch.
// ---------------------------------------------------------------------------
__global__ __launch_bounds__(256)
void k_conv3x3_wmma(const __bf16* __restrict__ Wbf,  // (256, 2304) row-major
                    const __bf16* __restrict__ Xt,   // (HW, 256) pixel-major
                    float* __restrict__ D,           // (256, H, W)
                    int H, int Wd)
{
  const int HW   = H * Wd;
  const int wave = threadIdx.x >> 5;
  const int lane = threadIdx.x & 31;
  const int n0 = (blockIdx.x * 8 + wave) << 4;   // pixel tile
  const int m0 = (int)blockIdx.y << 4;           // cout tile
  if (n0 >= HW) return;
  const int half = lane >> 4;
  const int l16  = lane & 15;
  const __bf16* arow = Wbf + (long)(m0 + l16) * 2304;
  const int bp = n0 + l16;                       // pixel this lane holds
  const int py = bp / Wd, px = bp - py * Wd;
  const __bf16 zero = f2bf(0.f);
  v8f acc = {};
  for (int k0 = 0; k0 < 2304; k0 += 32) {
    v8bf a0 = *(const v8bf*)(arow + k0 + (half << 3));
    v8bf a1 = *(const v8bf*)(arow + k0 + (half << 3) + 16);
    v16bf a = __builtin_shufflevector(a0, a1, 0,1,2,3,4,5,6,7,8,9,10,11,12,13,14,15);
    v16bf b;
#pragma unroll
    for (int e = 0; e < 16; ++e) {
      const int k  = k0 + (half << 3) + (e & 7) + ((e >> 3) << 4);       // k = ci*9 + ky*3 + kx
      const int ci = k / 9, rem = k - ci * 9;
      const int ky = rem / 3, kx = rem - ky * 3;
      const int yy = py + ky - 1, xx = px + kx - 1;
      __bf16 v = zero;
      if (yy >= 0 && yy < H && xx >= 0 && xx < Wd)
        v = Xt[(long)(yy * Wd + xx) * 256 + ci];
      b[e] = v;
    }
    acc = __builtin_amdgcn_wmma_f32_16x16x32_bf16(false, a, false, b,
                                                  (short)0, acc, false, false);
  }
#pragma unroll
  for (int r = 0; r < 8; ++r) {
    const int cm = m0 + r + (half << 3);         // cout
    D[(long)cm * HW + (n0 + l16)] = acc[r];
  }
}

// ---------------------------------------------------------------------------
// GroupNorm stats: one block per (b,group); 8 contiguous channels per group.
// ---------------------------------------------------------------------------
__global__ __launch_bounds__(256)
void k_gn_stats(const float* __restrict__ X, int HW, float* __restrict__ stats)
{
  __shared__ float s1[256], s2[256];
  const int idx = blockIdx.x;
  const int b = idx >> 5, g = idx & 31;
  const long base = ((long)b * 256 + g * 8) * HW;
  const int n = 8 * HW;
  float sum = 0.f, sq = 0.f;
  for (int t = threadIdx.x; t < n; t += 256) {
    float v = X[base + t];
    sum += v; sq += v * v;
  }
  s1[threadIdx.x] = sum; s2[threadIdx.x] = sq;
  __syncthreads();
  for (int s = 128; s > 0; s >>= 1) {
    if ((int)threadIdx.x < s) {
      s1[threadIdx.x] += s1[threadIdx.x + s];
      s2[threadIdx.x] += s2[threadIdx.x + s];
    }
    __syncthreads();
  }
  if (threadIdx.x == 0) {
    float m = s1[0] / n;
    float var = s2[0] / n - m * m;
    stats[idx]      = m;
    stats[64 + idx] = rsqrtf(var + 1e-5f);
  }
}

__global__ __launch_bounds__(256)
void k_gn_apply(const float* __restrict__ X, const float* __restrict__ stats,
                const float* __restrict__ gam, const float* __restrict__ bet,
                float* __restrict__ D, long o_bs, long o_ps, long o_cs,
                int HW, int relu, int total)
{
  int i = blockIdx.x * 256 + threadIdx.x;
  if (i >= total) return;
  const int pix = i % HW;
  const int c   = (i / HW) & 255;
  const int b   = i / (HW * 256);
  const int g   = b * 32 + (c >> 3);
  float v = (X[i] - stats[g]) * stats[64 + g] * gam[c] + bet[c];
  if (relu) v = fmaxf(v, 0.f);
  D[(long)b * o_bs + (long)pix * o_ps + (long)c * o_cs] = v;
}

// Sine positional embedding + level embed; (5376,256), shared by batch.
__global__ __launch_bounds__(256)
void k_pos_embed(float* __restrict__ pos, const float* __restrict__ lvl_emb)
{
  const int n = blockIdx.x, c = threadIdx.x;
  int lvl, H, W, pix;
  if (n < 256)       { lvl = 0; H = 16; W = 16; pix = n; }
  else if (n < 1280) { lvl = 1; H = 32; W = 32; pix = n - 256; }
  else               { lvl = 2; H = 64; W = 64; pix = n - 1280; }
  const int gy = pix / W, gx = pix - gy * W;
  const float TWO_PI = 6.283185307179586f;
  float p; int j;
  if (c < 128) { p = (gy + 1) / ((float)H + 1e-6f) * TWO_PI; j = c; }
  else         { p = (gx + 1) / ((float)W + 1e-6f) * TWO_PI; j = c - 128; }
  const float freq = powf(10000.0f, (float)(2 * (j >> 1)) * (1.0f / 128.0f));
  const float arg  = p / freq;
  const float v = (j & 1) ? cosf(arg) : sinf(arg);
  pos[(long)n * 256 + c] = v + lvl_emb[lvl * 256 + c];
}

// q = bf16(src + pos) directly (q only feeds WMMA GEMMs)
__global__ __launch_bounds__(256)
void k_add_pos(const float* __restrict__ src, const float* __restrict__ pos,
               __bf16* __restrict__ q, int total)
{
  int i = blockIdx.x * 256 + threadIdx.x;
  if (i >= total) return;
  q[i] = f2bf(src[i] + pos[i % 1376256]);
}

// Softmax over 12 = N_LEVELS*N_POINTS per (b,n,head), in place.
__global__ __launch_bounds__(256)
void k_softmax12(float* __restrict__ attn, int rows)
{
  int i = blockIdx.x * 256 + threadIdx.x;
  if (i >= rows) return;
  float* p = attn + (long)i * 12;
  float mx = p[0];
#pragma unroll
  for (int t = 1; t < 12; ++t) mx = fmaxf(mx, p[t]);
  float e[12], s = 0.f;
#pragma unroll
  for (int t = 0; t < 12; ++t) { e[t] = __expf(p[t] - mx); s += e[t]; }
  const float inv = 1.f / s;
#pragma unroll
  for (int t = 0; t < 12; ++t) p[t] = e[t] * inv;
}

// ---------------------------------------------------------------------------
// Multi-scale deformable attention sampling. Block per (b,n); thread=(head,d).
// ---------------------------------------------------------------------------
__global__ __launch_bounds__(256)
void k_msda_sample(const float* __restrict__ v, const float* __restrict__ off,
                   const float* __restrict__ attn, float* __restrict__ out)
{
  const int bn = blockIdx.x;
  const int b = bn / 5376, n = bn - b * 5376;
  const int h = threadIdx.x >> 5;
  const int d = threadIdx.x & 31;
  int Hq, Wq, pq;
  if (n < 256)       { Hq = 16; Wq = 16; pq = n; }
  else if (n < 1280) { Hq = 32; Wq = 32; pq = n - 256; }
  else               { Hq = 64; Wq = 64; pq = n - 1280; }
  const float refx = ((pq % Wq) + 0.5f) / Wq;
  const float refy = ((pq / Wq) + 0.5f) / Hq;
  const float* offp  = off  + (long)bn * 192 + h * 24;
  const float* attnp = attn + (long)bn * 96  + h * 12;
  const float* vb    = v + (long)b * 5376 * 256;
  float acc = 0.f;
  int start = 0, Hl = 16;
#pragma unroll
  for (int l = 0; l < 3; ++l) {
    const int Wl = Hl;
    const float fW = (float)Wl, fH = (float)Hl;
#pragma unroll
    for (int p = 0; p < 4; ++p) {
      const float ax = offp[l * 8 + p * 2 + 0];
      const float ay = offp[l * 8 + p * 2 + 1];
      const float aw = attnp[l * 4 + p];
      const float x = (refx + ax / fW) * fW - 0.5f;
      const float y = (refy + ay / fH) * fH - 0.5f;
      const float x0 = floorf(x), y0 = floorf(y);
      const float wx = x - x0, wy = y - y0;
      float s = 0.f;
#pragma unroll
      for (int t = 0; t < 4; ++t) {
        const float xx = x0 + (t & 1);
        const float yy = y0 + (t >> 1);
        if (xx >= 0.f && xx <= fW - 1.f && yy >= 0.f && yy <= fH - 1.f) {
          const int xi = (int)xx, yi = (int)yy;
          const float wgt = ((t & 1) ? wx : 1.f - wx) * ((t >> 1) ? wy : 1.f - wy);
          s += wgt * vb[(long)(start + yi * Wl + xi) * 256 + h * 32 + d];
        }
      }
      acc += aw * s;
    }
    start += Hl * Hl;
    Hl <<= 1;
  }
  out[(long)bn * 256 + h * 32 + d] = acc;
}

// LayerNorm over C=256; block per row; LDS tree reduction.
__global__ __launch_bounds__(256)
void k_layernorm(const float* __restrict__ X, const float* __restrict__ g,
                 const float* __restrict__ bta, float* __restrict__ D)
{
  __shared__ float s1[256], s2[256];
  const long r = blockIdx.x;
  const float v = X[r * 256 + threadIdx.x];
  s1[threadIdx.x] = v; s2[threadIdx.x] = v * v;
  __syncthreads();
  for (int s = 128; s > 0; s >>= 1) {
    if ((int)threadIdx.x < s) {
      s1[threadIdx.x] += s1[threadIdx.x + s];
      s2[threadIdx.x] += s2[threadIdx.x + s];
    }
    __syncthreads();
  }
  const float m    = s1[0] * (1.f / 256.f);
  const float var  = s2[0] * (1.f / 256.f) - m * m;
  const float rstd = rsqrtf(var + 1e-5f);
  D[r * 256 + threadIdx.x] = (v - m) * rstd * g[threadIdx.x] + bta[threadIdx.x];
}

// (B,N,C) slice -> NCHW output region.
__global__ __launch_bounds__(256)
void k_transpose_out(const float* __restrict__ src, float* __restrict__ dst,
                     int HW, int start, int total)
{
  int i = blockIdx.x * 256 + threadIdx.x;
  if (i >= total) return;
  const int pix = i % HW;
  const int c   = (i / HW) & 255;
  const int b   = i / (HW * 256);
  dst[i] = src[((long)(b * 5376 + start + pix)) * 256 + c];
}

// Half-pixel bilinear 64->128 upsample of `low`, added to `cur`.
__global__ __launch_bounds__(256)
void k_upsample_add(const float* __restrict__ cur, const float* __restrict__ low,
                    float* __restrict__ dst, int total)
{
  int i = blockIdx.x * 256 + threadIdx.x;
  if (i >= total) return;
  const int x = i & 127;
  const int y = (i >> 7) & 127;
  const long bc = (long)(i >> 14);
  const float sx = (x + 0.5f) * 0.5f - 0.5f;
  const float sy = (y + 0.5f) * 0.5f - 0.5f;
  const float fx0 = floorf(sx), fy0 = floorf(sy);
  const float wx = sx - fx0, wy = sy - fy0;
  int x0 = (int)fx0, y0 = (int)fy0, x1 = x0 + 1, y1 = y0 + 1;
  x0 = x0 < 0 ? 0 : (x0 > 63 ? 63 : x0);
  x1 = x1 < 0 ? 0 : (x1 > 63 ? 63 : x1);
  y0 = y0 < 0 ? 0 : (y0 > 63 ? 63 : y0);
  y1 = y1 < 0 ? 0 : (y1 > 63 ? 63 : y1);
  const float* p = low + bc * 4096;
  const float v00 = p[y0 * 64 + x0], v01 = p[y0 * 64 + x1];
  const float v10 = p[y1 * 64 + x0], v11 = p[y1 * 64 + x1];
  const float up = v00 * (1.f - wx) * (1.f - wy) + v01 * wx * (1.f - wy)
                 + v10 * (1.f - wx) * wy + v11 * wx * wy;
  dst[i] = cur[i] + up;
}

// ---------------------------------------------------------------------------
// Host orchestration.  setup_inputs() pytree flatten order (sorted dict keys):
// ---------------------------------------------------------------------------
#define I_F0 0
#define I_F1 1
#define I_F2 2
#define I_F3 3
#define I_FPN_LAT_GN_B 4
#define I_FPN_LAT_GN_G 5
#define I_FPN_LAT_W    6
#define I_FPN_OUT_GN_B 7
#define I_FPN_OUT_GN_G 8
#define I_FPN_OUT_W    9
#define I_PROJ(i, f)   (10 + (i) * 4 + (f))  // f: 0=b 1=gn_b 2=gn_g 3=w
#define I_LAYER(L)     (22 + (L) * 16)
// 0 W1, 1 W2, 2 Wattn, 3 Woff, 4 Wout, 5 Wv, 6 battn, 7 boff, 8 bout, 9 bv,
// 10 b1, 11 b2, 12 ln1_b, 13 ln1_g, 14 ln2_b, 15 ln2_g
#define I_LEVEL_EMBED 118
#define I_MASK_B      119
#define I_MASK_W      120

extern "C" void kernel_launch(void* const* d_in, const int* in_sizes, int n_in,
                              void* d_out, int out_size, void* d_ws, size_t ws_size,
                              hipStream_t stream)
{
  (void)in_sizes; (void)n_in; (void)out_size; (void)ws_size;
  auto IN = [&](int i) { return (const float*)d_in[i]; };
  float* out = (float*)d_out;
  char*  ws  = (char*)d_ws;

  // ---- workspace layout (byte offsets, all 256B aligned) ----
  const size_t O_SRC   = 0;            // f32 (2,5376,256)
  const size_t O_POS   = 11010048;     // f32 (5376,256)
  const size_t O_Q     = 16515072;     // f32 (2,5376,256)  msda out
  const size_t O_V     = 27525120;     // f32 (2,5376,256)
  const size_t O_OFF   = 38535168;     // f32 (2,5376,192)
  const size_t O_ATT   = 46792704;     // f32 (2,5376,96)
  const size_t O_TMP   = 50921472;     // f32 (2,5376,256) pre-LN sum
  const size_t O_BIG   = 61931520;     // f32 ffn hidden / fpn bufA+bufB (64MB)
  const size_t O_STA   = 129040384;    // f32 GN stats
  const size_t O_WBF   = 129040896;    // bf16 converted weights
  const size_t O_SRCBF = 141066752;    // bf16 (2,5376,256)
  const size_t O_QBF   = 146571776;    // bf16 (2,5376,256)
  const size_t O_ABF   = 152076800;    // bf16 shared: ffn hidden / transposed imgs
  float *srcp = (float*)(ws + O_SRC), *posp = (float*)(ws + O_POS);
  float *qp = (float*)(ws + O_Q), *vp = (float*)(ws + O_V);
  float *offp = (float*)(ws + O_OFF), *attnp = (float*)(ws + O_ATT);
  float *tmpp = (float*)(ws + O_TMP), *big = (float*)(ws + O_BIG);
  float *statsp = (float*)(ws + O_STA);
  __bf16 *wbf = (__bf16*)(ws + O_WBF), *srcbf = (__bf16*)(ws + O_SRCBF);
  __bf16 *qbf = (__bf16*)(ws + O_QBF), *abf = (__bf16*)(ws + O_ABF);

  // bf16 weight pool element offsets
  const long LW_STRIDE = 729088;
  const long LW_WV = 0, LW_WOFF = 65536, LW_WATTN = 114688, LW_WOUT = 139264,
             LW_W1 = 204800, LW_W2 = 466944;
  const long O_PROJW[3] = {4374528, 4898816, 5160960};
  const long O_LATW = 5292032, O_MASKW = 5357568, O_OUTW = 5423104;

  auto cvt = [&](const float* X, __bf16* Y, int tot) {
    k_cvt_bf16<<<(tot / 4 + 255) / 256, 256, 0, stream>>>(X, Y, tot);
  };
  auto cvtT = [&](const float* X, __bf16* Y, int R, int C) {
    k_cvt_t_bf16<<<(R * C + 255) / 256, 256, 0, stream>>>(X, Y, R, C);
  };
  auto gemm = [&](const __bf16* A, const __bf16* Bt, const float* bias, int bias_m,
                  const float* res, long rms, long rns,
                  float* D, long dms, long dns, int M, int N, int K, int relu) {
    dim3 grid((unsigned)((N / 16 + 7) / 8), (unsigned)(M / 16));
    k_wmma_gemm_bf<<<grid, 256, 0, stream>>>(A, Bt, bias, bias_m, res, rms, rns,
                                             D, dms, dns, M, N, K, relu);
  };

  // ---- convert all weights to bf16 once (transposed to (N,K) where needed) ----
  for (int L = 0; L < 6; ++L) {
    const int a = I_LAYER(L);
    const long lb = (long)L * LW_STRIDE;
    cvtT(IN(a + 5), wbf + lb + LW_WV,    256, 256);    // Wv   (K,N)->(N,K)
    cvtT(IN(a + 3), wbf + lb + LW_WOFF,  256, 192);    // Woff
    cvtT(IN(a + 2), wbf + lb + LW_WATTN, 256, 96);     // Wattn
    cvtT(IN(a + 4), wbf + lb + LW_WOUT,  256, 256);    // Wout
    cvtT(IN(a + 0), wbf + lb + LW_W1,    256, 1024);   // W1
    cvtT(IN(a + 1), wbf + lb + LW_W2,    1024, 256);   // W2
  }
  const int lvlH[3] = {16, 32, 64};
  const int lvlCin[3] = {2048, 1024, 512};
  const int lvlFeat[3] = {I_F3, I_F2, I_F1};
  const int lvlStart[3] = {0, 256, 1280};
  for (int i = 0; i < 3; ++i)
    cvt(IN(I_PROJ(i, 3)), wbf + O_PROJW[i], 256 * lvlCin[i]);  // (O,I) row-major
  cvt(IN(I_FPN_LAT_W), wbf + O_LATW, 65536);
  cvt(IN(I_MASK_W),    wbf + O_MASKW, 65536);
  cvt(IN(I_FPN_OUT_W), wbf + O_OUTW, 589824);                  // (O, I*3*3)

  // ---- input projections: feats_t = [f3,f2,f1] -> src (B,5376,256) ----
  for (int i = 0; i < 3; ++i) {
    const int Hs = lvlH[i], HW = Hs * Hs, Cin = lvlCin[i];
    const float* f = IN(lvlFeat[i]);
    for (int b = 0; b < 2; ++b)
      cvtT(f + (long)b * Cin * HW, abf + (long)b * HW * Cin, Cin, HW);
    for (int b = 0; b < 2; ++b)    // D[co][pix] = projW[co][:] . Xt[pix][:]
      gemm(wbf + O_PROJW[i], abf + (long)b * HW * Cin, IN(I_PROJ(i, 0)), 1,
           nullptr, 0, 0, big + (long)b * 256 * HW, HW, 1, 256, HW, Cin, 0);
    k_gn_stats<<<64, 256, 0, stream>>>(big, HW, statsp);
    const int total = 2 * 256 * HW;
    k_gn_apply<<<(total + 255) / 256, 256, 0, stream>>>(
        big, statsp, IN(I_PROJ(i, 2)), IN(I_PROJ(i, 1)),
        srcp + (long)lvlStart[i] * 256, (long)5376 * 256, 256, 1, HW, 0, total);
  }
  k_pos_embed<<<5376, 256, 0, stream>>>(posp, IN(I_LEVEL_EMBED));

  // ---- 6 encoder layers ----
  for (int L = 0; L < 6; ++L) {
    const int a = I_LAYER(L);
    const long lb = (long)L * LW_STRIDE;
    cvt(srcp, srcbf, 2752512);
    k_add_pos<<<(2752512 + 255) / 256, 256, 0, stream>>>(srcp, posp, qbf, 2752512);
    gemm(srcbf, wbf + lb + LW_WV,   IN(a + 9), 0, nullptr, 0, 0, vp,   256, 1, 10752, 256, 256, 0);
    gemm(qbf,   wbf + lb + LW_WOFF, IN(a + 7), 0, nullptr, 0, 0, offp, 192, 1, 10752, 192, 256, 0);
    gemm(qbf,   wbf + lb + LW_WATTN,IN(a + 6), 0, nullptr, 0, 0, attnp, 96, 1, 10752,  96, 256, 0);
    k_softmax12<<<(10752 * 8 + 255) / 256, 256, 0, stream>>>(attnp, 10752 * 8);
    k_msda_sample<<<10752, 256, 0, stream>>>(vp, offp, attnp, qp);
    cvt(qp, qbf, 2752512);
    gemm(qbf, wbf + lb + LW_WOUT, IN(a + 8), 0, srcp, 256, 1, tmpp, 256, 1, 10752, 256, 256, 0);
    k_layernorm<<<10752, 256, 0, stream>>>(tmpp, IN(a + 13), IN(a + 12), srcp);
    cvt(srcp, srcbf, 2752512);
    gemm(srcbf, wbf + lb + LW_W1, IN(a + 10), 0, nullptr, 0, 0, big, 1024, 1, 10752, 1024, 256, 1);
    cvt(big, abf, 11010048);
    gemm(abf, wbf + lb + LW_W2, IN(a + 11), 0, srcp, 256, 1, tmpp, 256, 1, 10752, 256, 1024, 0);
    k_layernorm<<<10752, 256, 0, stream>>>(tmpp, IN(a + 15), IN(a + 14), srcp);
  }

  // ---- outs[0..2] (NCHW) ----
  float* outp[3] = { out + 8388608, out + 8519680, out + 9043968 };
  for (int l = 0; l < 3; ++l) {
    const int HW = lvlH[l] * lvlH[l];
    const int total = 2 * 256 * HW;
    k_transpose_out<<<(total + 255) / 256, 256, 0, stream>>>(
        srcp, outp[l], HW, lvlStart[l], total);
  }

  // ---- FPN ----
  float* bufA = big;
  float* bufB = big + 8388608;
  for (int b = 0; b < 2; ++b)
    cvtT(IN(I_F0) + (long)b * 256 * 16384, abf + (long)b * 16384 * 256, 256, 16384);
  for (int b = 0; b < 2; ++b)
    gemm(wbf + O_LATW, abf + (long)b * 16384 * 256, nullptr, 0, nullptr, 0, 0,
         bufA + (long)b * 256 * 16384, 16384, 1, 256, 16384, 256, 0);
  k_gn_stats<<<64, 256, 0, stream>>>(bufA, 16384, statsp);
  k_gn_apply<<<(8388608 + 255) / 256, 256, 0, stream>>>(
      bufA, statsp, IN(I_FPN_LAT_GN_G), IN(I_FPN_LAT_GN_B),
      bufB, (long)256 * 16384, 1, 16384, 16384, 0, 8388608);
  k_upsample_add<<<(8388608 + 255) / 256, 256, 0, stream>>>(bufB, outp[2], bufA, 8388608);
  for (int b = 0; b < 2; ++b)                         // pixel-major bf16 image
    cvtT(bufA + (long)b * 256 * 16384, abf + (long)b * 16384 * 256, 256, 16384);
  for (int b = 0; b < 2; ++b)
    k_conv3x3_wmma<<<dim3(128, 16), 256, 0, stream>>>(
        wbf + O_OUTW, abf + (long)b * 16384 * 256,
        bufB + (long)b * 256 * 16384, 128, 128);
  k_gn_stats<<<64, 256, 0, stream>>>(bufB, 16384, statsp);
  k_gn_apply<<<(8388608 + 255) / 256, 256, 0, stream>>>(
      bufB, statsp, IN(I_FPN_OUT_GN_G), IN(I_FPN_OUT_GN_B),
      bufA, (long)256 * 16384, 1, 16384, 16384, 1, 8388608);
  for (int b = 0; b < 2; ++b)
    cvtT(bufA + (long)b * 256 * 16384, abf + (long)b * 16384 * 256, 256, 16384);
  for (int b = 0; b < 2; ++b)
    gemm(wbf + O_MASKW, abf + (long)b * 16384 * 256, IN(I_MASK_B), 1,
         nullptr, 0, 0, out + (long)b * 256 * 16384, 16384, 1, 256, 16384, 256, 0);
}
